// GroupingLayer_38216618999919
// MI455X (gfx1250) — matrix operand order repaired and checked by reading, not compiled, not run on hardware
//
#include <hip/hip_runtime.h>
#include <hip/hip_bf16.h>
#include <math.h>

typedef __bf16 bf16;
typedef __attribute__((ext_vector_type(2)))  __bf16 v2bf;
typedef __attribute__((ext_vector_type(8)))  __bf16 v8bf;
typedef __attribute__((ext_vector_type(16))) __bf16 v16bf;
typedef __attribute__((ext_vector_type(8)))  float  v8f;
typedef __attribute__((ext_vector_type(4)))  float  v4f;
typedef __attribute__((ext_vector_type(2)))  float  v2f;

#define EPSV 1e-5f

// problem dims (fixed by the reference)
constexpr int Bn   = 4;
constexpr int Cin  = 256;
constexpr int Hf   = 128;          // full resolution H = W
constexpr int Cout = 512;
constexpr int HoD  = 64;           // Ho = Wo
constexpr int Tpix = HoD * HoD;    // 4096 output pixels per batch
constexpr int Fpix = Hf * Hf;      // 16384 full-res pixels per batch

// ---------------------------------------------------------------------------
// WMMA GEMM: Out[b][o][p] = sum_i W[o][i] * X[b][i][p] + bias[o]
// BM=64, BN=64, BK=32. 256 threads = 8 wave32; each wave -> two 16x16 C tiles.
// GATHER: im2col for the stride-2 2x2 patchify conv (i = ci*4 + kh*2 + kw).
// ---------------------------------------------------------------------------
template<int KDIM, int NPIX, bool GATHER, bool OUT_BF16>
__global__ __launch_bounds__(256)
void gemm_wmma(const float* __restrict__ W, const float* __restrict__ bias,
               const float* __restrict__ X, void* __restrict__ Out)
{
  __shared__ bf16 Alds[64][40];    // [m][k], row stride 80B (16B aligned)
  __shared__ bf16 Blds[64][40];    // transposed: [n][k]

  const int tid  = threadIdx.x;
  const int lane = tid & 31;
  const int wid  = tid >> 5;
  const int nBase = blockIdx.x * 64;
  const int mBase = blockIdx.y * 64;
  const int b     = blockIdx.z;

  const float* xb = X + (size_t)b * KDIM * NPIX;

  const int mi     = wid & 3;         // M sub-tile (16 rows)
  const int njBase = (wid >> 2) * 2;  // first of two N sub-tiles

  v8f acc[2] = { (v8f){}, (v8f){} };

  // cooperative-load mappings
  const int ar = tid >> 2, ac = (tid & 3) * 8;   // A: 64 rows x 32 cols
  const int kp = tid >> 4, cg = (tid & 15) * 4;  // B: 16 K-row pairs x 16 col groups

  const int half = lane >> 4;   // 0: lanes 0-15, 1: lanes 16-31
  const int lm   = lane & 15;

  for (int k0 = 0; k0 < KDIM; k0 += 32) {
    // ---- stage A tile (f32 -> bf16)
    {
      const float* src = W + (size_t)(mBase + ar) * KDIM + k0 + ac;
      if (k0 + 32 < KDIM) __builtin_prefetch(src + 32, 0, 3);
      v4f a0 = *(const v4f*)(src);
      v4f a1 = *(const v4f*)(src + 4);
      v8bf dst;
      dst[0]=(bf16)a0[0]; dst[1]=(bf16)a0[1]; dst[2]=(bf16)a0[2]; dst[3]=(bf16)a0[3];
      dst[4]=(bf16)a1[0]; dst[5]=(bf16)a1[1]; dst[6]=(bf16)a1[2]; dst[7]=(bf16)a1[3];
      *(v8bf*)&Alds[ar][ac] = dst;
    }
    // ---- stage B tile, transposed into LDS; K rows processed as pairs so
    //      each (n, k-pair) packs into one dword -> ds_store_b32
    {
      const int i0 = k0 + 2 * kp;                 // even K row of the pair
      if (!GATHER) {
        const float* src0 = xb + (size_t)i0 * NPIX + nBase + cg;
        const float* src1 = src0 + NPIX;
        if (k0 + 32 < KDIM) __builtin_prefetch(src0 + 32 * NPIX, 0, 3);
        v4f b0 = *(const v4f*)(src0);
        v4f b1 = *(const v4f*)(src1);
        #pragma unroll
        for (int j = 0; j < 4; ++j) {
          v2bf pk; pk[0] = (bf16)b0[j]; pk[1] = (bf16)b1[j];
          *(v2bf*)&Blds[cg + j][2 * kp] = pk;
        }
      } else {
        // im2col: i0 even -> pair (i0, i0+1) = (ci, kh, kw=0) and kw=1, which
        // are adjacent in memory -> contiguous float2 gathers.
        const int ci = i0 >> 2, kh = (i0 >> 1) & 1;
        const int ho = nBase >> 6;                // one 64-wide N tile == one row
        const float* src = xb + (size_t)ci * Fpix + (2 * ho + kh) * Hf;
        if (k0 + 32 < KDIM) __builtin_prefetch(src + 8 * Fpix, 0, 3);
        #pragma unroll
        for (int j = 0; j < 4; ++j) {
          v2f x2 = *(const v2f*)(src + 2 * (cg + j));
          v2bf pk; pk[0] = (bf16)x2[0]; pk[1] = (bf16)x2[1];
          *(v2bf*)&Blds[cg + j][2 * kp] = pk;
        }
      }
    }
    __syncthreads();

    // ---- A fragment (ISA 16-bit A 16x32 layout): lane half selects K phase
    v8bf alo = *(const v8bf*)&Alds[mi * 16 + lm][half * 8];
    v8bf ahi = *(const v8bf*)&Alds[mi * 16 + lm][half * 8 + 16];
    v16bf afrag = __builtin_shufflevector(alo, ahi,
        0,1,2,3,4,5,6,7,8,9,10,11,12,13,14,15);

    // ---- two B fragments / two WMMAs
    #pragma unroll
    for (int t = 0; t < 2; ++t) {
      const int nj = njBase + t;
      v8bf blo = *(const v8bf*)&Blds[nj * 16 + lm][half * 8];
      v8bf bhi = *(const v8bf*)&Blds[nj * 16 + lm][half * 8 + 16];
      v16bf bfrag = __builtin_shufflevector(blo, bhi,
          0,1,2,3,4,5,6,7,8,9,10,11,12,13,14,15);
      acc[t] = __builtin_amdgcn_wmma_f32_16x16x32_bf16(
          false, afrag, false, bfrag, (short)0, acc[t], false, false);
    }
    __syncthreads();
  }

  // ---- epilogue: bias add + store (f32 or bf16)
  #pragma unroll
  for (int t = 0; t < 2; ++t) {
    const int nj = njBase + t;
    #pragma unroll
    for (int r = 0; r < 8; ++r) {
      const int o = mBase + mi * 16 + half * 8 + r;   // C layout: VGPR r, half
      const int p = nBase + nj * 16 + lm;
      float v = acc[t][r] + bias[o];
      size_t idx = ((size_t)b * Cout + o) * NPIX + p;
      if (OUT_BF16) ((bf16*)Out)[idx]  = (bf16)v;
      else          ((float*)Out)[idx] = v;
    }
  }
}

// ---------------------------------------------------------------------------
// Channel LayerNorm over (B, Cout, Tpix). One thread per pixel, coalesced
// across pixels. add=0: out = LN(in); add=1: out += LN(in).
// ---------------------------------------------------------------------------
__global__ __launch_bounds__(256)
void ln_channel(const float* __restrict__ in, float* __restrict__ out,
                const float* __restrict__ g, const float* __restrict__ bta,
                int add)
{
  const int p  = blockIdx.x * 256 + threadIdx.x;   // 0..Bn*Tpix-1
  const int b  = p >> 12;
  const int pp = p & (Tpix - 1);
  const float* ib = in + ((size_t)b * Cout) * Tpix + pp;
  float s = 0.f, sq = 0.f;
  for (int c = 0; c < Cout; ++c) {
    float v = ib[(size_t)c * Tpix];
    s += v; sq += v * v;
  }
  const float mu  = s * (1.f / Cout);
  const float var = sq * (1.f / Cout) - mu * mu;
  const float rs  = rsqrtf(var + EPSV);
  float* ob = out + ((size_t)b * Cout) * Tpix + pp;
  for (int c = 0; c < Cout; ++c) {
    float v = (ib[(size_t)c * Tpix] - mu) * rs * g[c] + bta[c];
    if (add) ob[(size_t)c * Tpix] += v;
    else     ob[(size_t)c * Tpix]  = v;
  }
}

// ---------------------------------------------------------------------------
// att[b, 2ho+kh, 2wo+kw] = softmax_over_window( sum_f k[b,f,2ho+kh,2wo+kw] *
//                                               q[b,f,ho,wo] )
// One block per (ho, b): thread t -> kh = t/128, col c = t%128, wo = c/2.
// ---------------------------------------------------------------------------
__global__ __launch_bounds__(256)
void att_softmax(const bf16* __restrict__ Kf, const float* __restrict__ Q,
                 float* __restrict__ att)
{
  const int ho = blockIdx.x, b = blockIdx.y;
  const int t  = threadIdx.x;
  const int kh = t >> 7;
  const int c  = t & 127;
  const int wo = c >> 1;
  const bf16*  kb = Kf + (size_t)b * Cout * Fpix + (size_t)(2 * ho + kh) * Hf + c;
  const float* qb = Q  + (size_t)b * Cout * Tpix + (size_t)ho * HoD + wo;
  float acc = 0.f;
  for (int f = 0; f < Cout; ++f)
    acc += (float)kb[(size_t)f * Fpix] * qb[(size_t)f * Tpix];

  __shared__ float sm[2][128];
  sm[kh][c] = acc;
  __syncthreads();
  const float a0 = sm[0][2 * wo], a1 = sm[0][2 * wo + 1];
  const float a2 = sm[1][2 * wo], a3 = sm[1][2 * wo + 1];
  const float m  = fmaxf(fmaxf(a0, a1), fmaxf(a2, a3));
  const float den = __expf(a0 - m) + __expf(a1 - m) + __expf(a2 - m) + __expf(a3 - m);
  att[(size_t)b * Fpix + (size_t)(2 * ho + kh) * Hf + c] = __expf(acc - m) / den;
}

// ---------------------------------------------------------------------------
// update[c,pix] = sum_window v[c, win(pix)] * att[win(pix)];
// x_out[c,pix] += LN_channel(update)[c,pix]. One thread per pixel, 2 passes.
// ---------------------------------------------------------------------------
__global__ __launch_bounds__(256)
void update_ln_add(const bf16* __restrict__ Vf, const float* __restrict__ att,
                   const float* __restrict__ g, const float* __restrict__ bta,
                   float* __restrict__ xout)
{
  const int p  = blockIdx.x * 256 + threadIdx.x;
  const int b  = p >> 12;
  const int pp = p & (Tpix - 1);
  const int ho = pp >> 6, wo = pp & 63;
  const float* ab = att + (size_t)b * Fpix;
  const float a00 = ab[(2 * ho) * Hf + 2 * wo];
  const float a01 = ab[(2 * ho) * Hf + 2 * wo + 1];
  const float a10 = ab[(2 * ho + 1) * Hf + 2 * wo];
  const float a11 = ab[(2 * ho + 1) * Hf + 2 * wo + 1];
  const bf16* vb = Vf + (size_t)b * Cout * Fpix + (size_t)(2 * ho) * Hf + 2 * wo;

  float s = 0.f, sq = 0.f;
  for (int c = 0; c < Cout; ++c) {
    const bf16* r = vb + (size_t)c * Fpix;
    float u = a00 * (float)r[0] + a01 * (float)r[1]
            + a10 * (float)r[Hf] + a11 * (float)r[Hf + 1];
    s += u; sq += u * u;
  }
  const float mu  = s * (1.f / Cout);
  const float var = sq * (1.f / Cout) - mu * mu;
  const float rs  = rsqrtf(var + EPSV);
  float* ob = xout + ((size_t)b * Cout) * Tpix + pp;
  for (int c = 0; c < Cout; ++c) {
    const bf16* r = vb + (size_t)c * Fpix;
    float u = a00 * (float)r[0] + a01 * (float)r[1]
            + a10 * (float)r[Hf] + a11 * (float)r[Hf + 1];
    ob[(size_t)c * Tpix] += (u - mu) * rs * g[c] + bta[c];
  }
}

// ---------------------------------------------------------------------------
extern "C" void kernel_launch(void* const* d_in, const int* in_sizes, int n_in,
                              void* d_out, int out_size, void* d_ws, size_t ws_size,
                              hipStream_t stream)
{
  const float* x_in   = (const float*)d_in[0];
  const float* conv_w = (const float*)d_in[1];
  const float* conv_b = (const float*)d_in[2];
  const float* k_w    = (const float*)d_in[3];
  const float* k_b    = (const float*)d_in[4];
  const float* q_w    = (const float*)d_in[5];
  const float* q_b    = (const float*)d_in[6];
  const float* v_w    = (const float*)d_in[7];
  const float* v_b    = (const float*)d_in[8];
  const float* mlp_w  = (const float*)d_in[9];
  const float* mlp_b  = (const float*)d_in[10];
  const float* lnc_g  = (const float*)d_in[11];
  const float* lnc_b  = (const float*)d_in[12];
  const float* lnv_g  = (const float*)d_in[13];
  const float* lnv_b  = (const float*)d_in[14];
  const float* lnm_g  = (const float*)d_in[15];
  const float* lnm_b  = (const float*)d_in[16];
  float* x_out = (float*)d_out;

  char* ws = (char*)d_ws;
  const size_t kvElems = (size_t)Bn * Cout * Fpix;    // 33.5M elems
  bf16*  k_bf = (bf16*)ws;                             // 64 MiB
  bf16*  v_bf = (bf16*)(ws + kvElems * sizeof(bf16));  // 64 MiB
  float* tmp  = (float*)(ws + 2 * kvElems * sizeof(bf16));               // 32 MiB
  float* att  = (float*)(ws + 2 * kvElems * sizeof(bf16)
                            + (size_t)Bn * Cout * Tpix * sizeof(float)); // 256 KiB
  (void)ws_size; (void)in_sizes; (void)n_in; (void)out_size;

  const dim3 blk(256);
  const dim3 gKV(Fpix / 64, Cout / 64, Bn);
  const dim3 gT (Tpix / 64, Cout / 64, Bn);
  const dim3 gPix(Bn * Tpix / 256);
  const dim3 gAtt(HoD, Bn);

  // k, v projections at full resolution -> bf16 workspace
  gemm_wmma<Cin, Fpix, false, true ><<<gKV, blk, 0, stream>>>(k_w, k_b, x_in, k_bf);
  gemm_wmma<Cin, Fpix, false, true ><<<gKV, blk, 0, stream>>>(v_w, v_b, x_in, v_bf);
  // patchify conv (im2col GEMM, K=1024) -> tmp ; channel LN -> x_out
  gemm_wmma<Cin * 4, Tpix, true, false><<<gT, blk, 0, stream>>>(conv_w, conv_b, x_in, tmp);
  ln_channel<<<gPix, blk, 0, stream>>>(tmp, x_out, lnc_g, lnc_b, 0);

  for (int it = 0; it < 3; ++it) {
    gemm_wmma<Cout, Tpix, false, false><<<gT, blk, 0, stream>>>(q_w, q_b, x_out, tmp);
    att_softmax<<<gAtt, blk, 0, stream>>>(k_bf, tmp, att);
    update_ln_add<<<gPix, blk, 0, stream>>>(v_bf, att, lnv_g, lnv_b, x_out);
    gemm_wmma<Cout, Tpix, false, false><<<gT, blk, 0, stream>>>(mlp_w, mlp_b, x_out, tmp);
    ln_channel<<<gPix, blk, 0, stream>>>(tmp, x_out, lnm_g, lnm_b, 1);
  }
}
